// GWK_social_88038239633654
// MI455X (gfx1250) — matrix operand order repaired and checked by reading, not compiled, not run on hardware
//
#include <hip/hip_runtime.h>
#include <hip/hip_bf16.h>
#include <cstdint>

#define N_NODES 6144
#define F_IN    128
#define D_MID   128
#define D_HID   64
#define NHEADS  4
#define NGRAPH  12
#define GSIZE   512
#define D_TGT   16
#define LEAKY_A 0.2f
#define EPS_F   1e-9f
#define BPAD    40   // padded LDS row stride (halfs): 80B -> conflict-free b128 frags

typedef __attribute__((ext_vector_type(16))) _Float16 v16h;
typedef __attribute__((ext_vector_type(8)))  _Float16 v8h;
typedef __attribute__((ext_vector_type(8)))  float    v8f;

__device__ __forceinline__ float leaky_f(float x) { return x >= 0.f ? x : LEAKY_A * x; }
__device__ __forceinline__ float elu_f(float x)   { return x >  0.f ? x : (__expf(x) - 1.f); }

// ---------------------------------------------------------------------------
// Small prep kernels
// ---------------------------------------------------------------------------
__global__ __launch_bounds__(256) void cvt_f32_to_f16(const float* __restrict__ src,
                                                      _Float16* __restrict__ dst, int n) {
    int i = blockIdx.x * 256 + threadIdx.x;
    if (i < n) dst[i] = (_Float16)src[i];
}

// dst[c*R + r] = src[r*C + c]   (src is [R][C] row-major f32, dst is [C][R] f16)
__global__ __launch_bounds__(256) void transpose_to_f16(const float* __restrict__ src,
                                                        _Float16* __restrict__ dst,
                                                        int R, int C) {
    int i = blockIdx.x * 256 + threadIdx.x;
    if (i < R * C) {
        int c = i / R, r = i - c * R;
        dst[i] = (_Float16)src[r * C + c];
    }
}

// W0 is [H][D_MID][D_HID]; build stacked-transposed BT [256][128]:
// BT[n][k] = W0[n>>6][k][n&63]
__global__ __launch_bounds__(256) void build_w0t(const float* __restrict__ W0,
                                                 _Float16* __restrict__ dst) {
    int i = blockIdx.x * 256 + threadIdx.x; // over 256*128
    if (i < (NHEADS * D_HID) * D_MID) {
        int n = i / D_MID, k = i - n * D_MID;
        dst[i] = (_Float16)W0[(n >> 6) * (D_MID * D_HID) + k * D_HID + (n & 63)];
    }
}

// ---------------------------------------------------------------------------
// WMMA GEMM:  C[M,*] = A[M,K] (f16 row-major) @ B  where BT is [N][K] f16.
// Block = 256 threads = 8 waves; wave tile = 16 rows x 64 cols; block = 128 x 64.
// B tile (64x32 f16) is staged in LDS with the CDNA5 async copy
// (global_load_async_to_lds_b128, ASYNCcnt-tracked): no VGPR round trip, and
// the next tile's DMA overlaps the current tile's WMMAs. Double-buffered.
// blockIdx.z batches independent problems (attention heads) via strides.
// batch512: attention batching — B's K index offset by (row-graph)*512.
// mode: 0 = relu(c+bias[n]) -> f16 row-major
//       1 = c               -> f16 TRANSPOSED (outH[n*ldc + m]), packed b128 stores
//       2 = elu(c)          -> f16 row-major
//       3 = elu(c)          -> f32 row-major
// ---------------------------------------------------------------------------
__global__ __launch_bounds__(256) void wmma_gemm_kernel(
    const _Float16* __restrict__ A, int lda, size_t aStrideZ,
    const _Float16* __restrict__ BT, int ldb, size_t bStrideZ,
    int K, int batch512,
    float* __restrict__ outF, _Float16* __restrict__ outH, int ldc, size_t outOffZ,
    int mode, const float* __restrict__ bias) {

    __shared__ __align__(16) _Float16 Bs[2][64 * BPAD];

    int z = blockIdx.z;
    A  += (size_t)z * aStrideZ;
    BT += (size_t)z * bStrideZ;
    if (outH) outH += (size_t)z * outOffZ;
    if (outF) outF += (size_t)z * outOffZ;

    int tid  = threadIdx.x;
    int wave = tid >> 5;
    int lane = tid & 31;
    int l15  = lane & 15;
    int hi   = lane >> 4;
    int row0 = blockIdx.x * 128 + wave * 16;
    int n0   = blockIdx.y * 64;
    int bko  = batch512 ? (row0 & ~(GSIZE - 1)) : 0;

    // cooperative B loader: thread -> (row 0..63, 8-half chunk 0..3); 16B each
    int brow = tid >> 2;
    int bch  = (tid & 3) * 8;
    const _Float16* bsrc = BT + (size_t)(n0 + brow) * ldb + bko + bch;
    // wave-relative LDS byte addresses (generic LDS ptr truncates to LDS offset)
    unsigned lds0 = (unsigned)(uintptr_t)&Bs[0][brow * BPAD + bch];
    unsigned lds1 = (unsigned)(uintptr_t)&Bs[1][brow * BPAD + bch];
    unsigned long long gbase = (unsigned long long)(uintptr_t)bsrc;

    // A fragment base: lane row = row0 + (lane&15); K interleave per ISA 7.12.2:
    // lanes 0-15 hold K {0..7, 16..23}, lanes 16-31 hold K {8..15, 24..31}
    const _Float16* arow = A + (size_t)(row0 + l15) * lda + hi * 8;

    v8f acc[4] = {v8f{}, v8f{}, v8f{}, v8f{}};

    // prologue: async-DMA tile 0 into Bs[0]
    asm volatile("global_load_async_to_lds_b128 %0, %1, off"
                 :: "v"(lds0), "v"(gbase) : "memory");
    v8h alo = *(const v8h*)arow;
    v8h ahi = *(const v8h*)(arow + 16);
    asm volatile("s_wait_asynccnt 0x0" ::: "memory");
    __syncthreads();

    for (int k = 0; k < K; k += 32) {
        int cur = (k >> 5) & 1;
        v8h nalo = alo, nahi = ahi;
        if (k + 32 < K) {
            // async prefetch of next B tile into the other LDS buffer;
            // completes under the WMMAs below, waited before the barrier.
            unsigned dst = cur ? lds0 : lds1;
            unsigned long long src = gbase + (unsigned long long)(k + 32) * sizeof(_Float16);
            asm volatile("global_load_async_to_lds_b128 %0, %1, off"
                         :: "v"(dst), "v"(src) : "memory");
            nalo = *(const v8h*)(arow + k + 32);
            nahi = *(const v8h*)(arow + k + 48);
        }
        v16h afrag;
#pragma unroll
        for (int i = 0; i < 8; i++) { afrag[i] = alo[i]; afrag[i + 8] = ahi[i]; }

        const _Float16* Bc = &Bs[cur][0];
#pragma unroll
        for (int t = 0; t < 4; t++) {
            // B fragment: lane col n = t*16+(lane&15); K {0..7, 16..23} (lanes 0-15)
            // or {8..15, 24..31} (lanes 16-31), matching A's interleave.
            const _Float16* q = Bc + (t * 16 + l15) * BPAD + hi * 8;
            v8h b0  = *(const v8h*)q;
            v8h b1v = *(const v8h*)(q + 16);
            v16h bfrag;
#pragma unroll
            for (int i = 0; i < 8; i++) { bfrag[i] = b0[i]; bfrag[i + 8] = b1v[i]; }
            acc[t] = __builtin_amdgcn_wmma_f32_16x16x32_f16(
                false, afrag, false, bfrag, (short)0, acc[t], false, false);
        }
        alo = nalo;
        ahi = nahi;
        asm volatile("s_wait_asynccnt 0x0" ::: "memory");
        __syncthreads();
    }

    // C layout: VGPR v -> row = row0 + v + hi*8, col = n0 + t*16 + (lane&15)
    if (mode == 1) {
        // transposed store: per (t, lane) the 8 accum rows are consecutive m
        // for fixed n -> one packed 16B store
#pragma unroll
        for (int t = 0; t < 4; t++) {
            v8h pk;
#pragma unroll
            for (int v = 0; v < 8; v++) pk[v] = (_Float16)acc[t][v];
            int n = n0 + t * 16 + l15;
            *(v8h*)&outH[(size_t)n * ldc + row0 + hi * 8] = pk;
        }
    } else {
#pragma unroll
        for (int t = 0; t < 4; t++) {
#pragma unroll
            for (int v = 0; v < 8; v++) {
                int m = row0 + v + hi * 8;
                int n = n0 + t * 16 + l15;
                float val = acc[t][v];
                if (mode == 0) {
                    val = fmaxf(val + bias[n], 0.f);
                    outH[(size_t)m * ldc + n] = (_Float16)val;
                } else if (mode == 2) {
                    outH[(size_t)m * ldc + n] = (_Float16)elu_f(val);
                } else {
                    outF[(size_t)m * ldc + n] = elu_f(val);
                }
            }
        }
    }
}

// ---------------------------------------------------------------------------
// a1[h][n] = sum_e ftT[(h*hd+e)*N + n] * al[h*hd+e]  (and a2 with ar)
// ---------------------------------------------------------------------------
__global__ __launch_bounds__(256) void avec_kernel(
    const _Float16* __restrict__ ftT, const float* __restrict__ al,
    const float* __restrict__ ar, float* __restrict__ a1, float* __restrict__ a2,
    int nheads, int hd) {
    int n = blockIdx.x * 256 + threadIdx.x;
    if (n >= N_NODES) return;
    for (int h = 0; h < nheads; h++) {
        float s1 = 0.f, s2 = 0.f;
        for (int e = 0; e < hd; e++) {
            float f = (float)ftT[(size_t)(h * hd + e) * N_NODES + n];
            s1 += f * al[h * hd + e];
            s2 += f * ar[h * hd + e];
        }
        a1[h * N_NODES + n] = s1;
        a2[h * N_NODES + n] = s2;
    }
}

// global max of a2 per head (one block per head)
__global__ __launch_bounds__(256) void a2max_kernel(const float* __restrict__ a2,
                                                    float* __restrict__ a2max) {
    int h = blockIdx.x;
    __shared__ float red[256];
    float m = -3.0e38f;
    for (int i = threadIdx.x; i < N_NODES; i += 256)
        m = fmaxf(m, a2[h * N_NODES + i]);
    red[threadIdx.x] = m;
    __syncthreads();
    for (int s = 128; s > 0; s >>= 1) {
        if (threadIdx.x < (unsigned)s)
            red[threadIdx.x] = fmaxf(red[threadIdx.x], red[threadIdx.x + s]);
        __syncthreads();
    }
    if (threadIdx.x == 0) a2max[h] = red[0];
}

// ---------------------------------------------------------------------------
// Softmax scores: one wave per (row, head). Full-row max == leaky(a1[i]+max_j a2[j])
// because leaky-ReLU is monotone, matching the reference exactly.
// P[h][row][j] = exp(leaky(a1+a2)-M)*mask / (rowsum + eps), j over the 512-block.
// ---------------------------------------------------------------------------
__global__ __launch_bounds__(256) void scores_kernel(
    const float* __restrict__ a1, const float* __restrict__ a2,
    const float* __restrict__ a2max, const float* __restrict__ mask,
    _Float16* __restrict__ P) {
    int head = blockIdx.y;
    int wave = threadIdx.x >> 5;
    int lane = threadIdx.x & 31;
    int row  = blockIdx.x * 8 + wave;
    int g    = row >> 9;

    float a1i = a1[head * N_NODES + row];
    float Mi  = leaky_f(a1i + a2max[head]);
    const float* a2g  = a2 + head * N_NODES + g * GSIZE;
    const float* mrow = mask + (size_t)row * N_NODES + g * GSIZE;
    _Float16* prow = P + ((size_t)head * N_NODES + row) * GSIZE;

    float sv[16];
    float s = 0.f;
#pragma unroll
    for (int t = 0; t < 16; t++) {
        int j = t * 32 + lane;
        float e = __expf(leaky_f(a1i + a2g[j]) - Mi) * mrow[j];
        sv[t] = e;
        s += e;
    }
#pragma unroll
    for (int o = 16; o >= 1; o >>= 1) s += __shfl_xor(s, o, 32);
    float inv = 1.f / (s + EPS_F);
#pragma unroll
    for (int t = 0; t < 16; t++) prow[t * 32 + lane] = (_Float16)(sv[t] * inv);
}

// ---------------------------------------------------------------------------
// Per-graph pooling + MLP head: out[g] = relu(pool@W1+b1)@W2+b2
// One block (128 threads) per graph.
// ---------------------------------------------------------------------------
__global__ __launch_bounds__(128) void pool_mlp_kernel(
    const float* __restrict__ hf, const float* __restrict__ W1,
    const float* __restrict__ b1, const float* __restrict__ W2,
    const float* __restrict__ b2, float* __restrict__ out) {
    int g = blockIdx.x;
    int t = threadIdx.x;
    __shared__ float pooled[D_MID];
    __shared__ float y1[D_MID];
    float s = 0.f;
    for (int r = 0; r < GSIZE; r++)
        s += hf[(size_t)(g * GSIZE + r) * D_MID + t];
    pooled[t] = s;
    __syncthreads();
    float acc = b1[t];
    for (int c = 0; c < D_MID; c++) acc += pooled[c] * W1[c * D_MID + t];
    y1[t] = fmaxf(acc, 0.f);
    __syncthreads();
    if (t < D_TGT) {
        float o = b2[t];
        for (int c = 0; c < D_MID; c++) o += y1[c] * W2[c * D_TGT + t];
        out[g * D_TGT + t] = o;
    }
}

// ---------------------------------------------------------------------------
extern "C" void kernel_launch(void* const* d_in, const int* in_sizes, int n_in,
                              void* d_out, int out_size, void* d_ws, size_t ws_size,
                              hipStream_t stream) {
    const float* x    = (const float*)d_in[0];
    const float* mask = (const float*)d_in[1];
    // d_in[2] = batch (unused: graphs are contiguous 512-node blocks)
    const float* Wv  = (const float*)d_in[3];
    const float* bv  = (const float*)d_in[4];
    const float* W0  = (const float*)d_in[5];
    const float* al0 = (const float*)d_in[6];
    const float* ar0 = (const float*)d_in[7];
    const float* Wf  = (const float*)d_in[8];
    const float* alf = (const float*)d_in[9];
    const float* arf = (const float*)d_in[10];
    const float* W1  = (const float*)d_in[11];
    const float* b1  = (const float*)d_in[12];
    const float* W2  = (const float*)d_in[13];
    const float* b2  = (const float*)d_in[14];

    // ---- workspace layout (bytes, 256-aligned) ----
    char* base = (char*)d_ws;
    size_t off = 0;
    auto alloc = [&](size_t bytes) {
        char* p = base + off;
        off = (off + bytes + 255) & ~(size_t)255;
        return p;
    };
    _Float16* x16   = (_Float16*)alloc((size_t)N_NODES * F_IN * 2);
    _Float16* WvT   = (_Float16*)alloc((size_t)F_IN * D_MID * 2);
    _Float16* W0T   = (_Float16*)alloc((size_t)(NHEADS * D_HID) * D_MID * 2);
    _Float16* WfT   = (_Float16*)alloc((size_t)D_MID * (NHEADS * D_HID) * 2);
    _Float16* h16   = (_Float16*)alloc((size_t)N_NODES * D_MID * 2);
    _Float16* ftT0  = (_Float16*)alloc((size_t)(NHEADS * D_HID) * N_NODES * 2);
    float*    a1_0  = (float*)alloc((size_t)NHEADS * N_NODES * 4);
    float*    a2_0  = (float*)alloc((size_t)NHEADS * N_NODES * 4);
    float*    a2m0  = (float*)alloc(256);
    _Float16* P0    = (_Float16*)alloc((size_t)NHEADS * N_NODES * GSIZE * 2);
    _Float16* h2    = (_Float16*)alloc((size_t)N_NODES * (NHEADS * D_HID) * 2);
    _Float16* ftfT  = (_Float16*)alloc((size_t)D_MID * N_NODES * 2);
    float*    a1f   = (float*)alloc((size_t)N_NODES * 4);
    float*    a2f   = (float*)alloc((size_t)N_NODES * 4);
    float*    a2mf  = (float*)alloc(256);
    _Float16* Pf    = (_Float16*)alloc((size_t)N_NODES * GSIZE * 2);
    float*    hf32  = (float*)alloc((size_t)N_NODES * D_MID * 4);
    (void)ws_size;

    // ---- 1) precision prep ----
    cvt_f32_to_f16<<<(N_NODES * F_IN + 255) / 256, 256, 0, stream>>>(x, x16, N_NODES * F_IN);
    transpose_to_f16<<<(F_IN * D_MID + 255) / 256, 256, 0, stream>>>(Wv, WvT, F_IN, D_MID);
    build_w0t<<<(NHEADS * D_HID * D_MID + 255) / 256, 256, 0, stream>>>(W0, W0T);
    transpose_to_f16<<<(NHEADS * D_HID * D_MID + 255) / 256, 256, 0, stream>>>(
        Wf, WfT, NHEADS * D_HID, D_MID);

    // ---- 2) h = relu(x@Wv + bv) ----
    wmma_gemm_kernel<<<dim3(N_NODES / 128, D_MID / 64, 1), 256, 0, stream>>>(
        x16, F_IN, 0, WvT, F_IN, 0, F_IN, 0, nullptr, h16, D_MID, 0, 0, bv);

    // ---- 3) ft0 = h@W0all  (stored transposed [256][6144]) ----
    wmma_gemm_kernel<<<dim3(N_NODES / 128, (NHEADS * D_HID) / 64, 1), 256, 0, stream>>>(
        h16, D_MID, 0, W0T, D_MID, 0, D_MID, 0, nullptr, ftT0, N_NODES, 0, 1, nullptr);

    // ---- 4) attention scalars, layer 0 ----
    avec_kernel<<<N_NODES / 256, 256, 0, stream>>>(ftT0, al0, ar0, a1_0, a2_0, NHEADS, D_HID);
    a2max_kernel<<<NHEADS, 256, 0, stream>>>(a2_0, a2m0);
    scores_kernel<<<dim3(N_NODES / 8, NHEADS), 256, 0, stream>>>(a1_0, a2_0, a2m0, mask, P0);

    // ---- 5) h2[:, h*64:(h+1)*64] = elu(P_h @ ft_h): all heads in one launch (z) ----
    wmma_gemm_kernel<<<dim3(N_NODES / 128, 1, NHEADS), 256, 0, stream>>>(
        P0, GSIZE, (size_t)N_NODES * GSIZE,
        ftT0, N_NODES, (size_t)D_HID * N_NODES,
        GSIZE, 1, nullptr, h2, NHEADS * D_HID, D_HID, 2, nullptr);

    // ---- 6) final head: ftf = h2@Wf (transposed out), scores, elu(Pf@ftf) ----
    wmma_gemm_kernel<<<dim3(N_NODES / 128, D_MID / 64, 1), 256, 0, stream>>>(
        h2, NHEADS * D_HID, 0, WfT, NHEADS * D_HID, 0, NHEADS * D_HID, 0,
        nullptr, ftfT, N_NODES, 0, 1, nullptr);
    avec_kernel<<<N_NODES / 256, 256, 0, stream>>>(ftfT, alf, arf, a1f, a2f, 1, D_MID);
    a2max_kernel<<<1, 256, 0, stream>>>(a2f, a2mf);
    scores_kernel<<<dim3(N_NODES / 8, 1), 256, 0, stream>>>(a1f, a2f, a2mf, mask, Pf);
    wmma_gemm_kernel<<<dim3(N_NODES / 128, D_MID / 64, 1), 256, 0, stream>>>(
        Pf, GSIZE, 0, ftfT, N_NODES, 0, GSIZE, 1, hf32, nullptr, D_MID, 0, 3, nullptr);

    // ---- 7) pool + MLP ----
    pool_mlp_kernel<<<NGRAPH, 128, 0, stream>>>(hf32, W1, b1, W2, b2, (float*)d_out);
}
